// FusionClassifier_38001870635070
// MI455X (gfx1250) — matrix operand order repaired and checked by reading, not compiled, not run on hardware
//
#include <hip/hip_runtime.h>
#include <hip/hip_bf16.h>

#define NN 100000
#define NE 1600000
#define NB 64
#define IND 78
#define HD 128
#define SD 768
#define FD 256

typedef __bf16 bf16_t;
typedef __attribute__((ext_vector_type(16))) __bf16 v16bf;
typedef __attribute__((ext_vector_type(8)))  __bf16 v8bf;
typedef __attribute__((ext_vector_type(8)))  float  v8f;
typedef __attribute__((ext_vector_type(4)))  int    v4i;

#define LDA 144  // LDS pitch in halfs (288B: 16B-aligned rows, bank-spread)
#define LDB 144

// ---- gfx1250 async global->LDS staging (guarded; falls back to manual) ------
#if defined(__has_builtin)
# if __has_builtin(__builtin_amdgcn_global_load_async_to_lds_b128)
#  define USE_ASYNC_LDS 1
# endif
#endif
#ifndef USE_ASYNC_LDS
# define USE_ASYNC_LDS 0
#endif

__device__ __forceinline__ void wait_async0() {
#if defined(__has_builtin)
# if __has_builtin(__builtin_amdgcn_s_wait_asynccnt)
  __builtin_amdgcn_s_wait_asynccnt(0);
  return;
# endif
#endif
  asm volatile("s_wait_asynccnt 0x0" ::: "memory");
}

#if USE_ASYNC_LDS
__device__ __forceinline__ void async_cp16(const bf16_t* g, bf16_t* l) {
  __builtin_amdgcn_global_load_async_to_lds_b128(
      (__attribute__((address_space(1))) v4i*)g,
      (__attribute__((address_space(3))) v4i*)l, 0, 0);
}
#endif

// ---------------- weight prep: Bt[n][k] = W[k][n], pad K to 128, f32->bf16 ----
__global__ __launch_bounds__(256) void prep_w(const float* __restrict__ W, int K,
                                              bf16_t* __restrict__ Bt) {
  int i = blockIdx.x * 256 + threadIdx.x;      // 128*128
  int n = i >> 7, k = i & 127;
  Bt[n * 128 + k] = (k < K) ? (bf16_t)W[k * 128 + n] : (bf16_t)0.0f;
}

// -------- pad x (N x 78) into h AND z (N x 128) f32 (z is the agg base) -----
__global__ __launch_bounds__(256) void pad_x(const float* __restrict__ x,
                                             float* __restrict__ h,
                                             float* __restrict__ z) {
  size_t i = (size_t)blockIdx.x * 256 + threadIdx.x;   // N*128
  int col = (int)(i & 127);
  size_t node = i >> 7;
  float v = (col < IND) ? x[node * IND + col] : 0.0f;
  h[i] = v;
  z[i] = v;
}

// ---------------- edge aggregation: z[dst] += h[src], one wave per edge ------
__global__ __launch_bounds__(256) void edge_agg(const float* __restrict__ h,
                                                const int* __restrict__ src,
                                                const int* __restrict__ dst,
                                                float* __restrict__ z, int ne) {
  int e = blockIdx.x * 8 + (threadIdx.x >> 5);
  if (e >= ne) return;
  int lane = threadIdx.x & 31;
  int s = src[e], d = dst[e];
  const float4 v = *(const float4*)(h + (size_t)s * 128 + lane * 4);
  float* zp = z + (size_t)d * 128 + lane * 4;
  unsafeAtomicAdd(zp + 0, v.x);
  unsafeAtomicAdd(zp + 1, v.y);
  unsafeAtomicAdd(zp + 2, v.z);
  unsafeAtomicAdd(zp + 3, v.w);
}

// ---------------- WMMA GEMM: [M x 128] * [128 x 128] -------------------------
// MODE 1: A f32 in global -> out = relu(A@W + bias) stored bf16
// MODE 2: A bf16 in global -> out = relu((A@W + bias)*bn_scale + bn_bias) f32,
//         dual-stored to out and (if non-null) out2 (next layer's agg base)
template <int MODE>
__global__ __launch_bounds__(256) void gin_gemm(const void* __restrict__ Aptr,
                                                const bf16_t* __restrict__ Bt,
                                                const float* __restrict__ bias,
                                                const float* __restrict__ bng,
                                                const float* __restrict__ bnb,
                                                void* __restrict__ outp,
                                                float* __restrict__ out2, int Mtot) {
  __shared__ bf16_t sA[128 * LDA];
  __shared__ bf16_t sB[128 * LDB];
  const int tid = threadIdx.x;
  const int wave = tid >> 5;
  const int lane = tid & 31;
  const int rowBase = blockIdx.x * 128;
  const int r = tid >> 1, c0 = (tid & 1) * 64;
  bool used_async = false;

  // stage Wt (128x128 bf16, row-major: [outcol][k])
#if USE_ASYNC_LDS
  {
    #pragma unroll
    for (int c = 0; c < 64; c += 8)
      async_cp16(Bt + r * 128 + c0 + c, &sB[r * LDB + c0 + c]);
    used_async = true;
  }
#else
  {
    #pragma unroll
    for (int c = 0; c < 64; c += 8) {
      v8bf vb = *(const v8bf*)(Bt + r * 128 + c0 + c);
      *(v8bf*)(&sB[r * LDB + c0 + c]) = vb;
    }
  }
#endif

  // stage A tile (128 rows x 128 k); MODE 1 converts f32->bf16 through VGPRs
  {
    int grow = rowBase + r;
    if (MODE == 1) {
      const float* A = (const float*)Aptr;
      for (int c = 0; c < 64; c += 4) {
        float4 f = make_float4(0.f, 0.f, 0.f, 0.f);
        if (grow < Mtot) f = *(const float4*)(A + (size_t)grow * 128 + c0 + c);
        bf16_t* d = &sA[r * LDA + c0 + c];
        d[0] = (bf16_t)f.x; d[1] = (bf16_t)f.y;
        d[2] = (bf16_t)f.z; d[3] = (bf16_t)f.w;
      }
    } else {
      const bf16_t* A = (const bf16_t*)Aptr;
#if USE_ASYNC_LDS
      if (rowBase + 128 <= Mtot) {      // full tile: direct async copy to LDS
        #pragma unroll
        for (int c = 0; c < 64; c += 8)
          async_cp16(A + (size_t)grow * 128 + c0 + c, &sA[r * LDA + c0 + c]);
        used_async = true;
      } else
#endif
      {
        for (int c = 0; c < 64; c += 8) {
          v8bf vb = {};
          if (grow < Mtot) vb = *(const v8bf*)(A + (size_t)grow * 128 + c0 + c);
          *(v8bf*)(&sA[r * LDA + c0 + c]) = vb;
        }
      }
    }
  }
  if (used_async) wait_async0();
  __syncthreads();

  const int r0 = wave * 16;        // this wave's 16 rows of the tile
  const int lrow = lane & 15;
  const int hi = lane >> 4;        // half-wave selector
  v8f zero = {};
  v8f acc[8];
  #pragma unroll
  for (int c = 0; c < 8; ++c) acc[c] = zero;

  #pragma unroll
  for (int kk = 0; kk < 128; kk += 32) {
    // A fragment: lanes 0-15 hold K {0..7,16..23}, lanes 16-31 {8..15,24..31}
    const bf16_t* ap = &sA[(r0 + lrow) * LDA + kk + hi * 8];
    v8bf a_lo = *(const v8bf*)(ap);
    v8bf a_hi = *(const v8bf*)(ap + 16);
    v16bf a;
    #pragma unroll
    for (int i = 0; i < 8; ++i) { a[i] = a_lo[i]; a[8 + i] = a_hi[i]; }
    #pragma unroll
    for (int c = 0; c < 8; ++c) {
      // B fragment: lane n = column, lanes 0-15 K {0..7,16..23}, lanes 16-31 K {8..15,24..31}
      const bf16_t* bp = &sB[(c * 16 + lrow) * LDB + kk + hi * 8];
      v8bf b_lo = *(const v8bf*)(bp);
      v8bf b_hi = *(const v8bf*)(bp + 16);
      v16bf b;
      #pragma unroll
      for (int i = 0; i < 8; ++i) { b[i] = b_lo[i]; b[8 + i] = b_hi[i]; }
      acc[c] = __builtin_amdgcn_wmma_f32_16x16x32_bf16(
          false, a, false, b, (short)0, acc[c], false, false);
    }
  }

  // epilogue: C/D layout: VGPR r, lanes 0-15 -> M=r, lanes 16-31 -> M=8+r
  #pragma unroll
  for (int c = 0; c < 8; ++c) {
    int col = c * 16 + lrow;
    float bv = bias[col];
    float gs = 0.f, bbv = 0.f;
    if (MODE == 2) { gs = bng[col] * rsqrtf(1.0f + 1e-5f); bbv = bnb[col]; }
    #pragma unroll
    for (int rr = 0; rr < 8; ++rr) {
      int grow = rowBase + r0 + hi * 8 + rr;
      if (grow < Mtot) {
        float v = acc[c][rr] + bv;
        if (MODE == 2) v = v * gs + bbv;
        v = fmaxf(v, 0.0f);
        if (MODE == 1) {
          ((bf16_t*)outp)[(size_t)grow * 128 + col] = (bf16_t)v;
        } else {
          ((float*)outp)[(size_t)grow * 128 + col] = v;
          if (out2) out2[(size_t)grow * 128 + col] = v;
        }
      }
    }
  }
}

// ---------------- pooling ----------------------------------------------------
__global__ __launch_bounds__(256) void pool_zero(float* sums, float* cnt) {
  int i = blockIdx.x * 256 + threadIdx.x;
  if (i < NB * 128) sums[i] = 0.f;
  if (i < NB) cnt[i] = 0.f;
}
__global__ __launch_bounds__(256) void pool_accum(const float* __restrict__ h,
                                                  const int* __restrict__ batch,
                                                  float* __restrict__ sums,
                                                  float* __restrict__ cnt) {
  size_t i = (size_t)blockIdx.x * 256 + threadIdx.x;   // N*32
  size_t node = i >> 5;
  int lane = (int)(i & 31);
  int b = batch[node];
  float4 v = *(const float4*)(h + node * 128 + lane * 4);
  float* sp = sums + (size_t)b * 128 + lane * 4;
  unsafeAtomicAdd(sp + 0, v.x);
  unsafeAtomicAdd(sp + 1, v.y);
  unsafeAtomicAdd(sp + 2, v.z);
  unsafeAtomicAdd(sp + 3, v.w);
  if (lane == 0) unsafeAtomicAdd(&cnt[b], 1.0f);
}
__global__ __launch_bounds__(256) void gf_div(const float* sums, const float* cnt,
                                              float* gf) {
  int i = blockIdx.x * 256 + threadIdx.x;   // 64*128
  gf[i] = sums[i] / fmaxf(cnt[i >> 7], 1.0f);
}

// ---------------- fusion head (tiny, scalar f32) -----------------------------
__global__ __launch_bounds__(256) void head_proj(const float* __restrict__ sem,
    const float* __restrict__ gf, const float* __restrict__ Ws, const float* __restrict__ bs,
    const float* __restrict__ Wg, const float* __restrict__ bg, float* __restrict__ tok) {
  int i = blockIdx.x * 256 + threadIdx.x;   // 64*256
  int b = i >> 8, f = i & 255;
  float as = bs[f];
  for (int k = 0; k < SD; ++k) as += sem[b * SD + k] * Ws[k * FD + f];
  float ag = bg[f];
  for (int k = 0; k < 128; ++k) ag += gf[b * 128 + k] * Wg[k * FD + f];
  tok[((size_t)b * 2 + 0) * FD + f] = as;
  tok[((size_t)b * 2 + 1) * FD + f] = ag;
}
__global__ __launch_bounds__(256) void head_qkv(const float* __restrict__ tok,
    const float* __restrict__ Wq, const float* __restrict__ Wk, const float* __restrict__ Wv,
    float* __restrict__ q, float* __restrict__ k, float* __restrict__ v) {
  int i = blockIdx.x * 256 + threadIdx.x;   // 64*2*256
  int bt = i >> 8, f = i & 255;
  float aq = 0.f, ak = 0.f, av = 0.f;
  for (int kk = 0; kk < FD; ++kk) {
    float t = tok[(size_t)bt * FD + kk];
    aq += t * Wq[kk * FD + f];
    ak += t * Wk[kk * FD + f];
    av += t * Wv[kk * FD + f];
  }
  q[(size_t)bt * FD + f] = aq;
  k[(size_t)bt * FD + f] = ak;
  v[(size_t)bt * FD + f] = av;
}
__global__ void head_att(const float* __restrict__ q, const float* __restrict__ k,
                         float* __restrict__ att) {
  int b = blockIdx.x * blockDim.x + threadIdx.x;
  if (b >= NB) return;
  float sc[2][2];
  for (int i = 0; i < 2; ++i)
    for (int j = 0; j < 2; ++j) {
      float d = 0.f;
      for (int f = 0; f < FD; ++f)
        d += q[((size_t)b * 2 + i) * FD + f] * k[((size_t)b * 2 + j) * FD + f];
      sc[i][j] = d * (1.0f / 16.0f);   // 1/sqrt(256)
    }
  for (int i = 0; i < 2; ++i) {
    float m = fmaxf(sc[i][0], sc[i][1]);
    float e0 = __expf(sc[i][0] - m), e1 = __expf(sc[i][1] - m);
    float s = e0 + e1;
    att[b * 4 + i * 2 + 0] = e0 / s;
    att[b * 4 + i * 2 + 1] = e1 / s;
  }
}
__global__ __launch_bounds__(256) void head_fuse(const float* __restrict__ att,
                                                 const float* __restrict__ v,
                                                 float* __restrict__ fused) {
  int i = blockIdx.x * 256 + threadIdx.x;   // 64*256
  int b = i >> 8, f = i & 255;
  float v0 = v[((size_t)b * 2 + 0) * FD + f];
  float v1 = v[((size_t)b * 2 + 1) * FD + f];
  float r = att[b * 4 + 0] * v0 + att[b * 4 + 1] * v1 +
            att[b * 4 + 2] * v0 + att[b * 4 + 3] * v1;
  fused[(size_t)b * FD + f] = 0.5f * r;
}
__global__ __launch_bounds__(256) void cls1(const float* __restrict__ fused,
                                            const float* __restrict__ Wc1,
                                            const float* __restrict__ bc1,
                                            float* __restrict__ hid) {
  int i = blockIdx.x * 256 + threadIdx.x;   // 64*128
  int b = i >> 7, j = i & 127;
  float a = bc1[j];
  for (int kk = 0; kk < FD; ++kk) a += fused[(size_t)b * FD + kk] * Wc1[kk * 128 + j];
  hid[(size_t)b * 128 + j] = fmaxf(a, 0.f);
}
__global__ void cls2(const float* __restrict__ hid, const float* __restrict__ Wc2,
                     const float* __restrict__ bc2, float* __restrict__ out) {
  int i = blockIdx.x * blockDim.x + threadIdx.x;   // 64*2
  if (i >= NB * 2) return;
  int b = i >> 1, o = i & 1;
  float a = bc2[o];
  for (int kk = 0; kk < 128; ++kk) a += hid[(size_t)b * 128 + kk] * Wc2[kk * 2 + o];
  out[b * 2 + o] = a;
}

// ---------------- workspace layout -------------------------------------------
static constexpr size_t OFF_H    = 0;
static constexpr size_t OFF_Z    = OFF_H + (size_t)NN * 128 * 4;
static constexpr size_t OFF_T    = OFF_Z + (size_t)NN * 128 * 4;
static constexpr size_t OFF_BT   = OFF_T + (size_t)NN * 128 * 2;
static constexpr size_t OFF_POOL = OFF_BT + (size_t)6 * 128 * 128 * 2;
static constexpr size_t OFF_CNT  = OFF_POOL + (size_t)NB * 128 * 4;
static constexpr size_t OFF_GF   = OFF_CNT + 256;
static constexpr size_t OFF_TOK  = OFF_GF + (size_t)NB * 128 * 4;
static constexpr size_t OFF_Q    = OFF_TOK + (size_t)NB * 2 * FD * 4;
static constexpr size_t OFF_K    = OFF_Q + (size_t)NB * 2 * FD * 4;
static constexpr size_t OFF_V    = OFF_K + (size_t)NB * 2 * FD * 4;
static constexpr size_t OFF_ATT  = OFF_V + (size_t)NB * 2 * FD * 4;
static constexpr size_t OFF_FUS  = OFF_ATT + 1024;
static constexpr size_t OFF_HID  = OFF_FUS + (size_t)NB * FD * 4;

extern "C" void kernel_launch(void* const* d_in, const int* in_sizes, int n_in,
                              void* d_out, int out_size, void* d_ws, size_t ws_size,
                              hipStream_t stream) {
  const float* sem   = (const float*)d_in[0];
  const float* x     = (const float*)d_in[1];
  const int*   ei    = (const int*)d_in[2];
  const int*   batch = (const int*)d_in[3];
  const float* W1[3] = {(const float*)d_in[4],  (const float*)d_in[8],  (const float*)d_in[12]};
  const float* b1[3] = {(const float*)d_in[5],  (const float*)d_in[9],  (const float*)d_in[13]};
  const float* W2[3] = {(const float*)d_in[6],  (const float*)d_in[10], (const float*)d_in[14]};
  const float* b2[3] = {(const float*)d_in[7],  (const float*)d_in[11], (const float*)d_in[15]};
  const float* bng[3] = {(const float*)d_in[16], (const float*)d_in[18], (const float*)d_in[20]};
  const float* bnb[3] = {(const float*)d_in[17], (const float*)d_in[19], (const float*)d_in[21]};
  const float* Ws = (const float*)d_in[22]; const float* bs = (const float*)d_in[23];
  const float* Wg = (const float*)d_in[24]; const float* bg = (const float*)d_in[25];
  const float* Wq = (const float*)d_in[26]; const float* Wk = (const float*)d_in[27];
  const float* Wv = (const float*)d_in[28];
  const float* Wc1 = (const float*)d_in[29]; const float* bc1 = (const float*)d_in[30];
  const float* Wc2 = (const float*)d_in[31]; const float* bc2 = (const float*)d_in[32];

  char* ws = (char*)d_ws;
  float*  h    = (float*)(ws + OFF_H);
  float*  z    = (float*)(ws + OFF_Z);
  bf16_t* t    = (bf16_t*)(ws + OFF_T);
  bf16_t* Bt   = (bf16_t*)(ws + OFF_BT);
  float*  sums = (float*)(ws + OFF_POOL);
  float*  cnt  = (float*)(ws + OFF_CNT);
  float*  gf   = (float*)(ws + OFF_GF);
  float*  tok  = (float*)(ws + OFF_TOK);
  float*  q    = (float*)(ws + OFF_Q);
  float*  kk   = (float*)(ws + OFF_K);
  float*  v    = (float*)(ws + OFF_V);
  float*  att  = (float*)(ws + OFF_ATT);
  float*  fus  = (float*)(ws + OFF_FUS);
  float*  hid  = (float*)(ws + OFF_HID);

  const int* srcv = ei;
  const int* dstv = ei + NE;
  const int gemmBlocks = (NN + 127) / 128;   // 782

  // transpose+pad+convert the 6 MLP weights (K = 78 for layer-1 W1, else 128)
  prep_w<<<64, 256, 0, stream>>>(W1[0], IND, Bt + 0 * 16384);
  prep_w<<<64, 256, 0, stream>>>(W2[0], 128, Bt + 1 * 16384);
  prep_w<<<64, 256, 0, stream>>>(W1[1], 128, Bt + 2 * 16384);
  prep_w<<<64, 256, 0, stream>>>(W2[1], 128, Bt + 3 * 16384);
  prep_w<<<64, 256, 0, stream>>>(W1[2], 128, Bt + 4 * 16384);
  prep_w<<<64, 256, 0, stream>>>(W2[2], 128, Bt + 5 * 16384);

  pad_x<<<(NN * 128) / 256, 256, 0, stream>>>(x, h, z);   // h0 = z0 = pad(x)

  for (int l = 0; l < 3; ++l) {
    edge_agg<<<(NE + 7) / 8, 256, 0, stream>>>(h, srcv, dstv, z, NE);  // z += agg(h)
    gin_gemm<1><<<gemmBlocks, 256, 0, stream>>>(z, Bt + (2 * l) * 16384,
        b1[l], nullptr, nullptr, t, nullptr, NN);        // t = relu(z@W1+b1) bf16
    gin_gemm<2><<<gemmBlocks, 256, 0, stream>>>(t, Bt + (2 * l + 1) * 16384,
        b2[l], bng[l], bnb[l], h, (l < 2) ? z : nullptr, NN);
    // h = relu(bn(t@W2+b2)) f32, dup into z as next layer's agg base
  }

  pool_zero<<<33, 256, 0, stream>>>(sums, cnt);
  pool_accum<<<(NN * 32) / 256, 256, 0, stream>>>(h, batch, sums, cnt);
  gf_div<<<(NB * 128) / 256, 256, 0, stream>>>(sums, cnt, gf);

  head_proj<<<(NB * FD) / 256, 256, 0, stream>>>(sem, gf, Ws, bs, Wg, bg, tok);
  head_qkv<<<(NB * 2 * FD) / 256, 256, 0, stream>>>(tok, Wq, Wk, Wv, q, kk, v);
  head_att<<<1, 64, 0, stream>>>(q, kk, att);
  head_fuse<<<(NB * FD) / 256, 256, 0, stream>>>(att, v, fus);
  cls1<<<(NB * 128) / 256, 256, 0, stream>>>(fus, Wc1, bc1, hid);
  cls2<<<1, 128, 0, stream>>>(hid, Wc2, bc2, (float*)d_out);
}
